// SimpleLSTM_79044578116023
// MI455X (gfx1250) — matrix operand order repaired and checked by reading, not compiled, not run on hardware
//
#include <hip/hip_runtime.h>

namespace {
constexpr int kSeq   = 48;   // timesteps
constexpr int kIn    = 64;   // input feature dim (K of projection GEMM)
constexpr int kHid   = 10;   // hidden size (K of recurrence GEMM)
constexpr int kFcLen = kSeq * kHid;  // 480
}

typedef __attribute__((ext_vector_type(2)))  float    v2f;
typedef __attribute__((ext_vector_type(4)))  float    v4f;
typedef __attribute__((ext_vector_type(8)))  float    v8f;
typedef __attribute__((ext_vector_type(16))) _Float16 v16h;

__device__ __forceinline__ float sigm_fast(float v) {
  return 1.0f / (1.0f + __expf(-v));
}
__device__ __forceinline__ float tanh_fast(float v) {
  return 1.0f - 2.0f / (__expf(2.0f * v) + 1.0f);
}

// -----------------------------------------------------------------------------
// Fully fused LSTM: one wave owns 16 batch rows for all 48 timesteps; x is
// streamed from HBM exactly once (the roofline floor), no workspace.
//
// Gates are computed TRANSPOSED, split into 4 gate-type 16x16 tiles (i,f,g,o):
//   D_tau[u, b] = sum_k W_ih[tau*10+u, k] * x[b, t, k]   (f16 WMMA, K=64)
//               + sum_k W_hh_ext[tau][u, k] * hext[b, k] (f32 WMMA, K=12)
// where hext[b, 10] == 1.0 and W_hh_ext[tau][u, 10] == (b_ih+b_hh)[tau*10+u]:
// the bias rides in the K-padding, so C starts from inline 0 (no init movs).
//
// C-fragment layout: lane (half = L>>4, n = L&15) VGPR j holds (u = j+8*half,
// batch n) -- identical for all four tiles, so the LSTM elementwise update is
// lane-aligned across the i/f/g/o accumulators.
//
// Padding: units u in [10,16) have all-zero A rows (incl. bias column) in both
// GEMMs -> gates 0 -> c stays 0 -> h stays 0 exactly; slot u=10 is re-stored
// as the constant 1.0 each step.
//
// h lives in a 16x16 LDS tile [batch][unit]; DS ops are in-order within a
// wave, so load(h_{t-1}) -> wmma -> store(h_t) needs no barrier.
// -----------------------------------------------------------------------------
__global__ __launch_bounds__(32) void lstm_fused(
    const float* __restrict__ x,
    const float* __restrict__ W_ih,
    const float* __restrict__ W_hh,
    const float* __restrict__ b_ih,
    const float* __restrict__ b_hh,
    const float* __restrict__ W_fc,
    const float* __restrict__ b_fc,
    float* __restrict__ out, int B)
{
  __shared__ float sWfc[2 * kFcLen];   // 960 floats
  __shared__ float h_sh[16 * 16];      // [batch][unit]; u=10 holds 1.0
  __shared__ float red[64];            // FC half-lane reduction

  const int L     = threadIdx.x;       // 0..31
  const int half  = L >> 4;
  const int n     = L & 15;            // batch column (N) / A row (M)
  const int bBase = blockIdx.x << 4;

  for (int i = L; i < 2 * kFcLen; i += 32) sWfc[i] = W_fc[i];
  for (int i = L; i < 256; i += 32)
    h_sh[i] = ((i & 15) == kHid) ? 1.0f : 0.0f;   // hext[b][10] = 1.0

  // ---- loop-invariant A fragments (register resident) ----
  const int   u_row = n;               // A-matrix row held by this lane
  const bool  okr   = (u_row < kHid);
  const float mrow  = okr ? 1.0f : 0.0f;

  // f16 A frags (projection, 16x32 layout): lanes 0-15 K=0..15, 16-31 K=16..31
  v16h aih[4][2];
#pragma unroll
  for (int tau = 0; tau < 4; ++tau) {
    const float* wr = W_ih + (size_t)(tau * kHid + min(u_row, kHid - 1)) * kIn;
#pragma unroll
    for (int s = 0; s < 2; ++s) {
      const int kb = 32 * s + 16 * half;
#pragma unroll
      for (int i = 0; i < 8; ++i) {
        aih[tau][s][i]     = (_Float16)(wr[kb + i] * mrow);
        aih[tau][s][8 + i] = (_Float16)(wr[kb + 8 + i] * mrow);
      }
    }
  }

  // f32 A frags (recurrence, 16x4 layout): lanes 0-15 K={0,1}, 16-31 K={2,3}.
  // Column k=10 carries the combined bias (paired with hext[b][10]==1).
  v2f ahh[4][3];
#pragma unroll
  for (int tau = 0; tau < 4; ++tau) {
    const int   g      = tau * kHid + min(u_row, kHid - 1);
    const float bias_g = okr ? (b_ih[g] + b_hh[g]) : 0.0f;
    const float* wr    = W_hh + (size_t)g * kHid;
#pragma unroll
    for (int s = 0; s < 3; ++s) {
      const int k = 4 * s + 2 * half;
      ahh[tau][s][0] = (k < kHid) ? (wr[k] * mrow)
                                  : ((k == kHid) ? bias_g : 0.0f);
      ahh[tau][s][1] = (k + 1 < kHid) ? (wr[k + 1] * mrow) : 0.0f;
    }
  }

  const float* xrow = x + (size_t)(bBase + n) * kSeq * kIn;

  float cst[8];
#pragma unroll
  for (int j = 0; j < 8; ++j) cst[j] = 0.0f;
  float acc0 = 0.0f, acc1 = 0.0f;

  __syncthreads();  // single-wave workgroup: S_NOP, kept for clarity

  // preload t=0 x fragment halves (raw f32, nontemporal; convert at use)
  v4f xq[8];
#pragma unroll
  for (int i = 0; i < 4; ++i) {
    xq[i]     = __builtin_nontemporal_load((const v4f*)(xrow + 16 * half + 4 * i));
    xq[4 + i] = __builtin_nontemporal_load((const v4f*)(xrow + 32 + 16 * half + 4 * i));
  }

#pragma unroll 2   // double-buffer via register renaming (kills xq copies)
  for (int t = 0; t < kSeq; ++t) {
    // issue next timestep's x loads early (independent of recurrence)
    const int tn = (t + 1 < kSeq) ? (t + 1) : t;
    v4f xqn[8];
#pragma unroll
    for (int i = 0; i < 4; ++i) {
      xqn[i]     = __builtin_nontemporal_load(
          (const v4f*)(xrow + (size_t)tn * kIn + 16 * half + 4 * i));
      xqn[4 + i] = __builtin_nontemporal_load(
          (const v4f*)(xrow + (size_t)tn * kIn + 32 + 16 * half + 4 * i));
    }

    // f16 B frags (x^T, 32x16 layout): lane = column b, K split by lane half
    v16h xb[2];
#pragma unroll
    for (int s = 0; s < 2; ++s)
#pragma unroll
      for (int i = 0; i < 16; ++i)
        xb[s][i] = (_Float16)xq[4 * s + (i >> 2)][i & 3];

    // f32 B' frags (hext^T, 4x16 layout), from LDS (h of step t-1)
    v2f hb[3];
#pragma unroll
    for (int s = 0; s < 3; ++s) {
      const int k = 4 * s + 2 * half;
      hb[s][0] = h_sh[n * 16 + k];
      hb[s][1] = h_sh[n * 16 + k + 1];
    }

    // gates^T = W_ih*x^T (f16 WMMA, C=inline 0) + W_hh_ext*hext^T (f32 WMMA)
    const v8f czero = {0.f, 0.f, 0.f, 0.f, 0.f, 0.f, 0.f, 0.f};
    v8f gacc[4];
#pragma unroll
    for (int tau = 0; tau < 4; ++tau) {
      gacc[tau] = __builtin_amdgcn_wmma_f32_16x16x32_f16(
          false, aih[tau][0], false, xb[0], (short)0, czero, false, false);
      gacc[tau] = __builtin_amdgcn_wmma_f32_16x16x32_f16(
          false, aih[tau][1], false, xb[1], (short)0, gacc[tau], false, false);
#pragma unroll
      for (int s = 0; s < 3; ++s)
        gacc[tau] = __builtin_amdgcn_wmma_f32_16x16x4_f32(
            false, ahh[tau][s], false, hb[s], (short)0, gacc[tau], false, false);
    }

    // lane-aligned LSTM update: lane owns units u=j+8*half of batch n
#pragma unroll
    for (int j = 0; j < 8; ++j) {
      const int u  = j + 8 * half;
      const float ig = sigm_fast(gacc[0][j]);
      const float fg = sigm_fast(gacc[1][j]);
      const float gg = tanh_fast(gacc[2][j]);
      const float og = sigm_fast(gacc[3][j]);
      const float cv = fmaf(fg, cst[j], ig * gg);
      cst[j] = cv;
      const float hv = og * tanh_fast(cv);
      h_sh[n * 16 + u] = (u == kHid) ? 1.0f : hv;    // keep bias lane at 1.0
      // fused FC: hv == 0 for padded units, so a clamped index is safe
      const int fidx = min(t * kHid + u, kFcLen - 1);
      acc0 = fmaf(sWfc[fidx], hv, acc0);
      acc1 = fmaf(sWfc[kFcLen + fidx], hv, acc1);
    }

#pragma unroll
    for (int i = 0; i < 8; ++i) xq[i] = xqn[i];
  }

  // combine FC partials of the two lane halves, add bias, store b64 per batch
  red[L * 2 + 0] = acc0;
  red[L * 2 + 1] = acc1;
  __syncthreads();
  if (L < 16) {
    v2f ov;
    ov[0] = red[L * 2]     + red[(L + 16) * 2]     + b_fc[0];
    ov[1] = red[L * 2 + 1] + red[(L + 16) * 2 + 1] + b_fc[1];
    *(v2f*)(out + (size_t)(bBase + L) * 2) = ov;
  }
}

// -----------------------------------------------------------------------------
extern "C" void kernel_launch(void* const* d_in, const int* in_sizes, int n_in,
                              void* d_out, int out_size, void* d_ws, size_t ws_size,
                              hipStream_t stream) {
  (void)n_in; (void)out_size; (void)d_ws; (void)ws_size;
  const float* x    = (const float*)d_in[0];
  const float* W_ih = (const float*)d_in[1];
  const float* W_hh = (const float*)d_in[2];
  const float* b_ih = (const float*)d_in[3];
  const float* b_hh = (const float*)d_in[4];
  const float* W_fc = (const float*)d_in[5];
  const float* b_fc = (const float*)d_in[6];
  float* out = (float*)d_out;

  const int B = in_sizes[0] / (kSeq * kIn);   // 8192

  lstm_fused<<<B / 16, 32, 0, stream>>>(x, W_ih, W_hh, b_ih, b_hh,
                                        W_fc, b_fc, out, B);
}
